// InformerModel_7799660609839
// MI455X (gfx1250) — compile-verified, matrix operand order checked by reading, not run on hardware
//
#include <hip/hip_runtime.h>
#include <math.h>

#define B_    4
#define L_    2048
#define H_    8
#define D_    64
#define SK_   38
#define NTOP_ 38
#define KPI   32            // keys per streaming iteration
#define NITER (L_ / KPI)

typedef _Float16 v16h __attribute__((ext_vector_type(16)));
typedef float    v8f  __attribute__((ext_vector_type(8)));
typedef int      v4i  __attribute__((vector_size(16)));

#if defined(__has_builtin)
#if __has_builtin(__builtin_amdgcn_global_load_async_to_lds_b128) && \
    __has_builtin(__builtin_amdgcn_s_wait_asynccnt)
#define HAVE_ASYNC_LDS 1
#endif
#endif

#define GPTR(p) ((__attribute__((address_space(1))) v4i*)(p))
#define LPTR(p) ((__attribute__((address_space(3))) v4i*)(p))

// ---------------------------------------------------------------------------
// Kernel 1: sparsity measure M[b,h,l] = max_k(q.k_samp) - sum_k(q.k_samp)/L
// ---------------------------------------------------------------------------
__global__ __launch_bounds__(256)
void mscore_kernel(const float* __restrict__ q, const float* __restrict__ k,
                   const int* __restrict__ samp, float* __restrict__ M)
{
    const int gid = blockIdx.x * blockDim.x + threadIdx.x;   // (b*H+h)*L + l
    const int l  = gid & (L_ - 1);
    const int bh = gid >> 11;
    const int h  = bh & (H_ - 1);
    const int b  = bh >> 3;

    const float4* qrow = (const float4*)(q + (((size_t)b * L_ + l) * H_ + h) * D_);
    float4 qv[16];
#pragma unroll
    for (int i = 0; i < 16; ++i) qv[i] = qrow[i];

    float mx = -3.0e38f, sm = 0.f;
    for (int s = 0; s < SK_; ++s) {
        const int idx = samp[l * SK_ + s];
        const float4* krow = (const float4*)(k + (((size_t)b * L_ + idx) * H_ + h) * D_);
        float acc = 0.f;
#pragma unroll
        for (int i = 0; i < 16; ++i) {
            float4 kv = krow[i];
            acc = fmaf(qv[i].x, kv.x, fmaf(qv[i].y, kv.y,
                  fmaf(qv[i].z, kv.z, fmaf(qv[i].w, kv.w, acc))));
        }
        mx = fmaxf(mx, acc);
        sm += acc;
    }
    M[gid] = mx - sm * (1.0f / (float)L_);
}

// ---------------------------------------------------------------------------
// Kernel 2: top-38 indices of M per (b,h); iterative argmax, ties -> low index
// ---------------------------------------------------------------------------
__global__ __launch_bounds__(256)
void topk_kernel(const float* __restrict__ M, int* __restrict__ Mtop)
{
    __shared__ float vals[L_];
    __shared__ float rv[256];
    __shared__ int   ri[256];

    const int bh  = blockIdx.x;
    const int tid = threadIdx.x;
    const float* m = M + (size_t)bh * L_;

    for (int i = tid; i < L_; i += 256) vals[i] = m[i];
    __syncthreads();

    for (int t = 0; t < NTOP_; ++t) {
        float bv = -3.0e38f; int bi = L_;
        for (int i = tid; i < L_; i += 256) {
            float v = vals[i];
            if (v > bv) { bv = v; bi = i; }
        }
        rv[tid] = bv; ri[tid] = bi;
        __syncthreads();
        for (int s = 128; s > 0; s >>= 1) {
            if (tid < s) {
                float ov = rv[tid + s]; int oi = ri[tid + s];
                if (ov > rv[tid] || (ov == rv[tid] && oi < ri[tid])) { rv[tid] = ov; ri[tid] = oi; }
            }
            __syncthreads();
        }
        if (tid == 0) { Mtop[bh * NTOP_ + t] = ri[0]; vals[ri[0]] = -3.0e38f; }
        __syncthreads();
    }
}

// ---------------------------------------------------------------------------
// Kernel 3: column mean of V per (b,h)
// ---------------------------------------------------------------------------
__global__ __launch_bounds__(64)
void vmean_kernel(const float* __restrict__ v, float* __restrict__ vmean)
{
    const int bh = blockIdx.x;
    const int d  = threadIdx.x;
    const int b  = bh >> 3, h = bh & 7;
    float s = 0.f;
    for (int l = 0; l < L_; ++l)
        s += v[(((size_t)b * L_ + l) * H_ + h) * D_ + d];
    vmean[bh * D_ + d] = s * (1.0f / (float)L_);
}

// ---------------------------------------------------------------------------
// Kernel 4: broadcast mean into the whole output (default context)
// ---------------------------------------------------------------------------
__global__ __launch_bounds__(256)
void fill_kernel(const float* __restrict__ vmean, float* __restrict__ out)
{
    const int i  = blockIdx.x * 256 + threadIdx.x;    // over B*L*H*D
    const int hd = i & (H_ * D_ - 1);
    const int b  = i >> 20;
    out[i] = vmean[b * (H_ * D_) + hd];
}

// ---------------------------------------------------------------------------
// Kernel 4b: f32 -> f16 precision/layout conversion for the WMMA streaming path
//   Kh  [bh][key][dim]  (per-(b,h) contiguous rows)
//   Vth [bh][dim][key]  (pre-transposed so B-fragments are contiguous)
// ---------------------------------------------------------------------------
__global__ __launch_bounds__(256)
void cvt_kernel(const float* __restrict__ k, const float* __restrict__ v,
                _Float16* __restrict__ Kh, _Float16* __restrict__ Vth)
{
    const int i = blockIdx.x * 256 + threadIdx.x;     // ((b*L+l)*H+h)*D+d
    const int d = i & 63;
    const int h = (i >> 6) & 7;
    const int l = (i >> 9) & (L_ - 1);
    const int b = i >> 20;
    const int bh = b * H_ + h;
    Kh [((size_t)bh * L_ + l) * D_ + d] = (_Float16)k[i];
    Vth[((size_t)bh * D_ + d) * L_ + l] = (_Float16)v[i];
}

// ---------------------------------------------------------------------------
// Kernel 5: dense attention of the 38 top queries vs all keys — f16 WMMA,
// flash-style online softmax, f32 accumulation. One block per (b,h), 3 waves.
// PRE=1: f16 pre-converted inputs + async-to-LDS double-buffered staging.
// PRE=0: fallback, f32 inputs converted during synchronous staging.
// ---------------------------------------------------------------------------
template <bool PRE>
__global__ __launch_bounds__(96)
void attn_kernel(const float* __restrict__ Qg, const float* __restrict__ Kg,
                 const float* __restrict__ Vg, const _Float16* __restrict__ Kh,
                 const _Float16* __restrict__ Vth, const int* __restrict__ Mtop,
                 float* __restrict__ out)
{
    __shared__ __align__(16) _Float16 KtB[2][KPI * D_];   // [key][dim]
    __shared__ __align__(16) _Float16 VtB[2][D_ * KPI];   // [dim][key]
    __shared__ __align__(16) _Float16 Pw[3][16 * KPI];    // per-wave P tile

    const int bh   = blockIdx.x;
    const int b    = bh >> 3, h = bh & 7;
    const int tid  = threadIdx.x;
    const int wave = tid >> 5;
    const int lane = tid & 31;
    const int lr   = lane & 15;
    const int hi   = lane >> 4;

    // ---- Q_reduce A-fragments (16 rows per wave, rows >= 38 clamped) ----
    const int  u    = wave * 16 + lr;
    const int  qrow = (u < NTOP_) ? Mtop[bh * NTOP_ + u] : Mtop[bh * NTOP_];
    const float* qp = Qg + (((size_t)b * L_ + qrow) * H_ + h) * D_;
    v16h qa[2];
#pragma unroll
    for (int c = 0; c < 2; ++c) {
        const int base = c * 32 + (hi ? 8 : 0);
#pragma unroll
        for (int e = 0; e < 8; ++e) {
            qa[c][e]     = (_Float16)qp[base + e];
            qa[c][8 + e] = (_Float16)qp[base + 16 + e];
        }
    }

    v16h ones;
#pragma unroll
    for (int e = 0; e < 16; ++e) ones[e] = (_Float16)1.0f;

    v8f  O[4];
    float rmax[8], rsum[8];
#pragma unroll
    for (int t = 0; t < 4; ++t)
#pragma unroll
        for (int r = 0; r < 8; ++r) O[t][r] = 0.f;
#pragma unroll
    for (int r = 0; r < 8; ++r) { rmax[r] = -3.0e38f; rsum[r] = 0.f; }

    const float scale = 0.125f;                 // 1/sqrt(64)
    _Float16* P = &Pw[wave][0];

    // ---- compute on one staged 32-key tile ----
    auto compute = [&](const _Float16* kt, const _Float16* vt) {
        v8f S[2];
#pragma unroll
        for (int s = 0; s < 2; ++s) {
            const int keyb = s * 16 + lr;
            v8f acc;
#pragma unroll
            for (int r = 0; r < 8; ++r) acc[r] = 0.f;
#pragma unroll
            for (int c = 0; c < 2; ++c) {
                v16h bf;
                const int base = c * 32 + (hi ? 8 : 0);
                const _Float16* kp = &kt[keyb * D_];
#pragma unroll
                for (int e = 0; e < 8; ++e) {
                    bf[e]     = kp[base + e];
                    bf[8 + e] = kp[base + 16 + e];
                }
                acc = __builtin_amdgcn_wmma_f32_16x16x32_f16(
                          false, qa[c], false, bf, (short)0, acc, false, false);
            }
#pragma unroll
            for (int r = 0; r < 8; ++r) acc[r] *= scale;
            S[s] = acc;
        }

        // online softmax: row max via 16-lane butterfly; row sum via ones-WMMA
        float scb[8], p0b[8], p1b[8];
#pragma unroll
        for (int r = 0; r < 8; ++r) {
            const float v0 = S[0][r], v1 = S[1][r];
            float mloc = fmaxf(v0, v1);
#pragma unroll
            for (int mk = 1; mk < 16; mk <<= 1)
                mloc = fmaxf(mloc, __shfl_xor(mloc, mk, 32));
            const float mnew = fmaxf(rmax[r], mloc);
            const float sc   = __expf(rmax[r] - mnew);
            p0b[r] = __expf(v0 - mnew);
            p1b[r] = __expf(v1 - mnew);
            rmax[r] = mnew;
            scb[r]  = sc;
#pragma unroll
            for (int t = 0; t < 4; ++t) O[t][r] *= sc;
        }

        // C-layout -> A-layout bounce through per-wave LDS
#pragma unroll
        for (int r = 0; r < 8; ++r) {
            const int m = r + 8 * hi;
            P[m * KPI + lr]      = (_Float16)p0b[r];
            P[m * KPI + 16 + lr] = (_Float16)p1b[r];
        }
        asm volatile("s_wait_dscnt 0" ::: "memory");   // DS in-order + compiler fence

        v16h pf;
        {
            const int base = hi ? 8 : 0;
            const _Float16* pp = &P[lr * KPI];
#pragma unroll
            for (int e = 0; e < 8; ++e) {
                pf[e]     = pp[base + e];
                pf[8 + e] = pp[base + 16 + e];
            }
        }

        // row sums of P: one WMMA against all-ones B (every column = rowsum)
        v8f zs;
#pragma unroll
        for (int r = 0; r < 8; ++r) zs[r] = 0.f;
        const v8f siter = __builtin_amdgcn_wmma_f32_16x16x32_f16(
                              false, pf, false, ones, (short)0, zs, false, false);
#pragma unroll
        for (int r = 0; r < 8; ++r) rsum[r] = rsum[r] * scb[r] + siter[r];

        // O += P(16x32) @ V(32x16) for 4 dim-tiles
#pragma unroll
        for (int t = 0; t < 4; ++t) {
            v16h vf;
            const int dim = t * 16 + lr;
            const _Float16* vp = &vt[dim * KPI];
            const int base = hi ? 8 : 0;
#pragma unroll
            for (int e = 0; e < 8; ++e) {
                vf[e]     = vp[base + e];
                vf[8 + e] = vp[base + 16 + e];
            }
            O[t] = __builtin_amdgcn_wmma_f32_16x16x32_f16(
                       false, pf, false, vf, (short)0, O[t], false, false);
        }
    };

    if constexpr (PRE) {
#ifdef HAVE_ASYNC_LDS
        // double-buffered async-to-LDS pipeline (ASYNCcnt-tracked DMA)
        auto stage = [&](int buf, int tile) {
            const int k0 = tile * KPI;
            const _Float16* ktile = Kh  + ((size_t)bh * L_ + k0) * D_;
            const _Float16* vtile = Vth + (size_t)bh * D_ * L_ + k0;
            for (int i = tid; i < 256; i += 96) {
                __builtin_amdgcn_global_load_async_to_lds_b128(
                    GPTR(ktile + (size_t)i * 8), LPTR(&KtB[buf][i * 8]), 0, 0);
                const int dv = i >> 2, cv = i & 3;
                __builtin_amdgcn_global_load_async_to_lds_b128(
                    GPTR(vtile + (size_t)dv * L_ + cv * 8),
                    LPTR(&VtB[buf][dv * KPI + cv * 8]), 0, 0);
            }
        };
        stage(0, 0);
        for (int it = 0; it < NITER; ++it) {
            const int cur = it & 1;
            __builtin_amdgcn_s_wait_asynccnt(0);   // my copies for 'cur' done
            __syncthreads();                       // everyone's copies done
            if (it + 1 < NITER) stage(cur ^ 1, it + 1);
            compute(&KtB[cur][0], &VtB[cur][0]);
        }
#else
        // synchronous f16 staging (16B vector copies)
        for (int it = 0; it < NITER; ++it) {
            const int k0 = it * KPI;
            const _Float16* ktile = Kh  + ((size_t)bh * L_ + k0) * D_;
            const _Float16* vtile = Vth + (size_t)bh * D_ * L_ + k0;
            __syncthreads();
            for (int i = tid; i < 128; i += 96)
                ((int4*)&KtB[0][0])[i] = ((const int4*)ktile)[i];
            for (int i = tid; i < 256; i += 96) {
                const int dv = i >> 2, cv = i & 3;
                ((int4*)&VtB[0][dv * KPI])[cv] =
                    ((const int4*)(vtile + (size_t)dv * L_))[cv];
            }
            __syncthreads();
            compute(&KtB[0][0], &VtB[0][0]);
        }
#endif
    } else {
        // fallback: stage from f32 inputs with on-the-fly conversion
        for (int it = 0; it < NITER; ++it) {
            const int k0 = it * KPI;
            __syncthreads();
            for (int i = tid; i < KPI * D_; i += 96) {
                const int key = i >> 6, dim = i & 63;
                const size_t g = (((size_t)b * L_ + (k0 + key)) * H_ + h) * D_ + dim;
                KtB[0][key * D_ + dim]  = (_Float16)Kg[g];
                VtB[0][dim * KPI + key] = (_Float16)Vg[g];
            }
            if (it + 1 < NITER && tid < KPI) {
                const size_t g = (((size_t)b * L_ + (k0 + KPI + tid)) * H_ + h) * D_;
                __builtin_prefetch(&Kg[g], 0, 1);
                __builtin_prefetch(&Vg[g], 0, 1);
            }
            __syncthreads();
            compute(&KtB[0][0], &VtB[0][0]);
        }
    }

    // ---- epilogue: normalize and scatter the 38 valid rows ----
#pragma unroll
    for (int r = 0; r < 8; ++r) {
        const int m  = r + 8 * hi;
        const int uu = wave * 16 + m;
        if (uu < NTOP_) {
            const int   row = Mtop[bh * NTOP_ + uu];
            const float inv = 1.0f / rsum[r];
#pragma unroll
            for (int t = 0; t < 4; ++t) {
                const int dim = t * 16 + lr;
                out[(((size_t)b * L_ + row) * H_ + h) * D_ + dim] = O[t][r] * inv;
            }
        }
    }
}

// ---------------------------------------------------------------------------
extern "C" void kernel_launch(void* const* d_in, const int* in_sizes, int n_in,
                              void* d_out, int out_size, void* d_ws, size_t ws_size,
                              hipStream_t stream)
{
    (void)in_sizes; (void)n_in; (void)out_size;
    const float* q    = (const float*)d_in[0];
    const float* k    = (const float*)d_in[1];
    const float* v    = (const float*)d_in[2];
    const int*   samp = (const int*)d_in[3];
    float*       out  = (float*)d_out;

    char* ws = (char*)d_ws;
    float* Mbuf  = (float*)ws;                          // 256 KB
    int*   Mtop  = (int*)(ws + 262144);                 // 32*38 int
    float* vmean = (float*)(ws + 270336);               // 8 KB
    _Float16* Kh  = (_Float16*)(ws + 278528);           // 8 MB
    _Float16* Vth = (_Float16*)(ws + 278528 + 8388608); // 8 MB
    const size_t need = 278528 + 2ull * 8388608;

    mscore_kernel<<<(B_ * H_ * L_) / 256, 256, 0, stream>>>(q, k, samp, Mbuf);
    topk_kernel  <<<B_ * H_, 256, 0, stream>>>(Mbuf, Mtop);
    vmean_kernel <<<B_ * H_, 64, 0, stream>>>(v, vmean);
    fill_kernel  <<<(B_ * L_ * H_ * D_) / 256, 256, 0, stream>>>(vmean, out);

    if (ws_size >= need) {
        cvt_kernel<<<(B_ * L_ * H_ * D_) / 256, 256, 0, stream>>>(k, v, Kh, Vth);
        attn_kernel<true><<<B_ * H_, 96, 0, stream>>>(q, k, v, Kh, Vth, Mtop, out);
    } else {
        attn_kernel<false><<<B_ * H_, 96, 0, stream>>>(q, k, v, Kh, Vth, Mtop, out);
    }
}
